// MaskingAwareAttention_82240033784118
// MI455X (gfx1250) — matrix operand order repaired and checked by reading, not compiled, not run on hardware
//
#include <hip/hip_runtime.h>

// ---- problem dims ----
#define BB 4
#define TT 2048
#define DD 512
#define HH 8
#define DK 64
#define NROWS (BB*TT)          // 8192
#define RELN (2*TT - 1)        // 4095

typedef __bf16 bf16_t;
typedef bf16_t v16bf __attribute__((ext_vector_type(16)));
typedef float  v8f   __attribute__((ext_vector_type(8)));
typedef unsigned int v4u __attribute__((ext_vector_type(4)));

union Frag { v16bf v; v4u u[2]; };

__device__ __forceinline__ unsigned short f2bf(float f) {
    unsigned int u = __float_as_uint(f);
    u += 0x7FFFu + ((u >> 16) & 1u);
    return (unsigned short)(u >> 16);
}

// ---------------- prep: x + sinusoidal PE -> bf16 ----------------
__global__ void prep_x_pe(const float* __restrict__ x, unsigned short* __restrict__ xpe) {
    int idx = blockIdx.x * blockDim.x + threadIdx.x;
    if (idx >= NROWS * DD) return;
    int d = idx & (DD - 1);
    int t = (idx >> 9) & (TT - 1);
    int d2 = d & ~1;  // 2*(d/2)
    float freq = __expf(-(float)d2 * (9.210340371976184f / (float)DD));
    float ang = (float)t * freq;
    float pe = (d & 1) ? __cosf(ang) : __sinf(ang);
    xpe[idx] = f2bf(x[idx] + pe);
}

// ---------------- prep: f32 -> bf16 (weights) ----------------
__global__ void f32_to_bf16(const float* __restrict__ in, unsigned short* __restrict__ out, int n) {
    int idx = blockIdx.x * blockDim.x + threadIdx.x;
    if (idx < n) out[idx] = f2bf(in[idx]);
}

// ---------------- prep: antisymmetrized relative bias ----------------
__global__ void prep_antibias(const float* __restrict__ E, float* __restrict__ out) {
    int idx = blockIdx.x * blockDim.x + threadIdx.x;
    if (idx >= RELN * HH) return;
    int h = idx & (HH - 1);
    int i = idx >> 3;
    out[idx] = 0.5f * (E[i * HH + h] - E[(RELN - 1 - i) * HH + h]);
}

// ---------------- projection GEMM: C[8192x512] = A x W^T + bias ----------------
// Register-blocked: one wave computes a 64x32 tile (4 M-tiles x 2 N-tiles).
// mode 0: Q = C*0.125 -> bf16 [b,h,t,dk]
// mode 1: K = C       -> bf16 [b,h,t,dk]
// mode 2: V = C       -> bf16 transposed [b,h,dk,t]
// mode 3: out = C     -> f32  [b,t,e]
__global__ __launch_bounds__(32) void proj_gemm(
    const unsigned short* __restrict__ A,   // [8192,512] bf16 row-major
    const unsigned short* __restrict__ Wb,  // [512,512]  bf16 row-major [e][d]
    const float* __restrict__ bias,         // [512]
    unsigned short* __restrict__ outB,
    float* __restrict__ outF,
    int mode)
{
    int mt = blockIdx.x;            // 0..127  (rows/64)
    int nt = blockIdx.y;            // 0..15   (cols/32)
    int lane = threadIdx.x;
    int lm = lane & 15;
    int lh = lane >> 4;
    int mbase = mt * 64;
    int nbase = nt * 32;

    const unsigned short* arow[4];
    const unsigned short* brow[2];
    v8f c[4][2];
#pragma unroll
    for (int ni = 0; ni < 2; ++ni) {
        brow[ni] = Wb + (size_t)(nbase + ni * 16 + lm) * DD;
        float bval = bias[nbase + ni * 16 + lm];
#pragma unroll
        for (int mi = 0; mi < 4; ++mi)
#pragma unroll
            for (int i = 0; i < 8; ++i) c[mi][ni][i] = bval;
    }
#pragma unroll
    for (int mi = 0; mi < 4; ++mi)
        arow[mi] = A + (size_t)(mbase + mi * 16 + lm) * DD;

#pragma unroll 1
    for (int kb = 0; kb < DD / 32; ++kb) {
        int k0 = kb * 32;
        Frag a[4], b[2];
#pragma unroll
        for (int mi = 0; mi < 4; ++mi) {
            const unsigned short* ap = arow[mi] + k0 + lh * 8;
            a[mi].u[0] = *(const v4u*)(ap);
            a[mi].u[1] = *(const v4u*)(ap + 16);
        }
#pragma unroll
        for (int ni = 0; ni < 2; ++ni) {
            const unsigned short* bp = brow[ni] + k0 + lh * 8;
            b[ni].u[0] = *(const v4u*)(bp);
            b[ni].u[1] = *(const v4u*)(bp + 16);
        }
#pragma unroll
        for (int mi = 0; mi < 4; ++mi)
#pragma unroll
            for (int ni = 0; ni < 2; ++ni)
                c[mi][ni] = __builtin_amdgcn_wmma_f32_16x16x32_bf16(
                    false, a[mi].v, false, b[ni].v, (short)0, c[mi][ni], false, false);
    }

#pragma unroll
    for (int mi = 0; mi < 4; ++mi) {
#pragma unroll
        for (int ni = 0; ni < 2; ++ni) {
            int n = nbase + ni * 16 + lm;
#pragma unroll
            for (int i = 0; i < 8; ++i) {
                int m = mbase + mi * 16 + lh * 8 + i;   // global row 0..8191
                float v = c[mi][ni][i];
                if (mode == 3) {
                    outF[(size_t)m * DD + n] = v;
                } else {
                    int b_ = m >> 11;
                    int t_ = m & (TT - 1);
                    int h_ = n >> 6;
                    int dk = n & (DK - 1);
                    size_t bh = (size_t)(b_ * HH + h_);
                    if (mode == 0)
                        outB[(bh * TT + t_) * DK + dk] = f2bf(v * 0.125f);
                    else if (mode == 1)
                        outB[(bh * TT + t_) * DK + dk] = f2bf(v);
                    else
                        outB[(bh * DK + dk) * TT + t_] = f2bf(v);
                }
            }
        }
    }
}

// ---------------- flash attention: one wave per (b,h,16-row q tile) ----------------
__global__ __launch_bounds__(32) void flash_attn(
    const unsigned short* __restrict__ Qb,   // [b,h,t,dk] bf16 (pre-scaled)
    const unsigned short* __restrict__ Kb,   // [b,h,t,dk] bf16
    const unsigned short* __restrict__ Vt,   // [b,h,dk,t] bf16
    const unsigned char*  __restrict__ mask, // [b,t] bool
    const float* __restrict__ antibias,      // [4095,8]
    unsigned short* __restrict__ AO)         // [b,t,h*64+dk] bf16
{
    __shared__ __align__(16) unsigned short Pl[16 * 32];

    int qt = blockIdx.x;      // 0..127
    int h  = blockIdx.y;      // 0..7
    int b  = blockIdx.z;      // 0..3
    int lane = threadIdx.x;
    int lm = lane & 15;
    int lh = lane >> 4;
    int qbase = qt * 16;
    size_t bh = (size_t)(b * HH + h);

    // Q A-fragments (k0 = 0 and k0 = 32), persistent for whole kv loop
    const unsigned short* qrow = Qb + (bh * TT + qbase + lm) * DK;
    Frag aq0, aq1;
    aq0.u[0] = *(const v4u*)(qrow + 0  + lh * 8);
    aq0.u[1] = *(const v4u*)(qrow + 16 + lh * 8);
    aq1.u[0] = *(const v4u*)(qrow + 32 + lh * 8);
    aq1.u[1] = *(const v4u*)(qrow + 48 + lh * 8);

    v8f o[4];
#pragma unroll
    for (int nt = 0; nt < 4; ++nt)
#pragma unroll
        for (int i = 0; i < 8; ++i) o[nt][i] = 0.f;

    float Mrow[8], Lrow[8];
#pragma unroll
    for (int i = 0; i < 8; ++i) { Mrow[i] = -INFINITY; Lrow[i] = 0.f; }

    const unsigned char* mrow = mask + (size_t)b * TT;
    const unsigned short* kbase_ptr = Kb + bh * TT * DK;
    const unsigned short* vbase_ptr = Vt + bh * DK * TT;

    for (int kb2 = 0; kb2 < TT / 32; ++kb2) {
        int kbase = kb2 * 32;

        // ---- S = Q K^T for 32 keys: two 16x16 C tiles ----
        v8f s0, s1;
#pragma unroll
        for (int i = 0; i < 8; ++i) { s0[i] = 0.f; s1[i] = 0.f; }
        {
            const unsigned short* kp0 = kbase_ptr + (size_t)(kbase + lm) * DK + lh * 8;
            Frag b00, b01;
            b00.u[0] = *(const v4u*)(kp0);       b00.u[1] = *(const v4u*)(kp0 + 16);
            b01.u[0] = *(const v4u*)(kp0 + 32);  b01.u[1] = *(const v4u*)(kp0 + 48);
            s0 = __builtin_amdgcn_wmma_f32_16x16x32_bf16(false, aq0.v, false, b00.v, (short)0, s0, false, false);
            s0 = __builtin_amdgcn_wmma_f32_16x16x32_bf16(false, aq1.v, false, b01.v, (short)0, s0, false, false);

            const unsigned short* kp1 = kbase_ptr + (size_t)(kbase + 16 + lm) * DK + lh * 8;
            Frag b10, b11;
            b10.u[0] = *(const v4u*)(kp1);       b10.u[1] = *(const v4u*)(kp1 + 16);
            b11.u[0] = *(const v4u*)(kp1 + 32);  b11.u[1] = *(const v4u*)(kp1 + 48);
            s1 = __builtin_amdgcn_wmma_f32_16x16x32_bf16(false, aq0.v, false, b10.v, (short)0, s1, false, false);
            s1 = __builtin_amdgcn_wmma_f32_16x16x32_bf16(false, aq1.v, false, b11.v, (short)0, s1, false, false);
        }

        // ---- mask + rel-pos bias + online softmax ----
        bool msk0 = mrow[kbase + lm] != 0;
        bool msk1 = mrow[kbase + 16 + lm] != 0;
        float p0[8], p1[8], scl[8];
#pragma unroll
        for (int i = 0; i < 8; ++i) {
            int q  = qbase + lh * 8 + i;
            int d0 = q - (kbase + lm) + (TT - 1);   // in [0,4094]
            float v0 = msk0 ? (s0[i] + antibias[d0 * HH + h])        : -INFINITY;
            float v1 = msk1 ? (s1[i] + antibias[(d0 - 16) * HH + h]) : -INFINITY;

            float mx = fmaxf(v0, v1);
#pragma unroll
            for (int off = 1; off < 16; off <<= 1)
                mx = fmaxf(mx, __shfl_xor(mx, off, 32));

            float Mn = fmaxf(Mrow[i], mx);
            float sc = (Mrow[i] == -INFINITY) ? 0.f : __expf(Mrow[i] - Mn);
            float e0 = msk0 ? __expf(v0 - Mn) : 0.f;
            float e1 = msk1 ? __expf(v1 - Mn) : 0.f;

            float srow = e0 + e1;
#pragma unroll
            for (int off = 1; off < 16; off <<= 1)
                srow += __shfl_xor(srow, off, 32);

            Lrow[i] = Lrow[i] * sc + srow;
            Mrow[i] = Mn;
            scl[i] = sc;
            p0[i] = e0;
            p1[i] = e1;
        }

        // rescale O accumulators (same (i,lh) -> row mapping as S tiles)
#pragma unroll
        for (int nt = 0; nt < 4; ++nt)
#pragma unroll
            for (int i = 0; i < 8; ++i) o[nt][i] *= scl[i];

        // ---- stage P through LDS: C layout -> A layout ----
        __syncthreads();
#pragma unroll
        for (int i = 0; i < 8; ++i) {
            int m = lh * 8 + i;
            Pl[m * 32 + lm]      = f2bf(p0[i]);
            Pl[m * 32 + 16 + lm] = f2bf(p1[i]);
        }
        __syncthreads();
        Frag pa;
        {
            const unsigned short* pr = &Pl[lm * 32];
            pa.u[0] = *(const v4u*)(pr + lh * 8);
            pa.u[1] = *(const v4u*)(pr + 16 + lh * 8);
        }

        // ---- O += P V ----
#pragma unroll
        for (int nt = 0; nt < 4; ++nt) {
            const unsigned short* vp = vbase_ptr + (size_t)(nt * 16 + lm) * TT + kbase + lh * 8;
            Frag bv;
            bv.u[0] = *(const v4u*)(vp);
            bv.u[1] = *(const v4u*)(vp + 16);
            o[nt] = __builtin_amdgcn_wmma_f32_16x16x32_bf16(false, pa.v, false, bv.v, (short)0, o[nt], false, false);
        }
    }

    // ---- finalize: O /= L, store bf16 [b,t,h*64+dk] ----
    float rl[8];
#pragma unroll
    for (int i = 0; i < 8; ++i) rl[i] = (Lrow[i] > 0.f) ? (1.0f / Lrow[i]) : 0.f;

#pragma unroll
    for (int nt = 0; nt < 4; ++nt) {
        int dk = nt * 16 + lm;
#pragma unroll
        for (int i = 0; i < 8; ++i) {
            int q = qbase + lh * 8 + i;
            AO[((size_t)(b * TT + q)) * DD + h * DK + dk] = f2bf(o[nt][i] * rl[i]);
        }
    }
}

// ---------------- host launcher ----------------
extern "C" void kernel_launch(void* const* d_in, const int* in_sizes, int n_in,
                              void* d_out, int out_size, void* d_ws, size_t ws_size,
                              hipStream_t stream) {
    const float*         x    = (const float*)d_in[0];
    const unsigned char* mask = (const unsigned char*)d_in[1];
    const float* Wq = (const float*)d_in[2];
    const float* bq = (const float*)d_in[3];
    const float* Wk = (const float*)d_in[4];
    const float* bk = (const float*)d_in[5];
    const float* Wv = (const float*)d_in[6];
    const float* bv = (const float*)d_in[7];
    const float* Wo = (const float*)d_in[8];
    const float* bo = (const float*)d_in[9];
    // d_in[10] missing_bias is all-zeros and only applied to -inf entries -> no-op
    const float* rel = (const float*)d_in[11];
    float* out = (float*)d_out;

    char* w = (char*)d_ws;
    unsigned short* xpe = (unsigned short*)w; w += (size_t)NROWS * DD * 2;   // 8 MB
    unsigned short* Wqb = (unsigned short*)w; w += (size_t)DD * DD * 2;
    unsigned short* Wkb = (unsigned short*)w; w += (size_t)DD * DD * 2;
    unsigned short* Wvb = (unsigned short*)w; w += (size_t)DD * DD * 2;
    unsigned short* Wob = (unsigned short*)w; w += (size_t)DD * DD * 2;
    unsigned short* Qb  = (unsigned short*)w; w += (size_t)NROWS * DD * 2;   // [b,h,t,dk]
    unsigned short* Kb  = (unsigned short*)w; w += (size_t)NROWS * DD * 2;
    unsigned short* Vt  = (unsigned short*)w; w += (size_t)NROWS * DD * 2;   // [b,h,dk,t]
    unsigned short* AO  = (unsigned short*)w; w += (size_t)NROWS * DD * 2;
    float* anti         = (float*)w;          w += (size_t)RELN * HH * 4;

    {
        int n = NROWS * DD;
        prep_x_pe<<<(n + 255) / 256, 256, 0, stream>>>(x, xpe);
    }
    {
        int n = DD * DD;
        int g = (n + 255) / 256;
        f32_to_bf16<<<g, 256, 0, stream>>>(Wq, Wqb, n);
        f32_to_bf16<<<g, 256, 0, stream>>>(Wk, Wkb, n);
        f32_to_bf16<<<g, 256, 0, stream>>>(Wv, Wvb, n);
        f32_to_bf16<<<g, 256, 0, stream>>>(Wo, Wob, n);
    }
    {
        int n = RELN * HH;
        prep_antibias<<<(n + 255) / 256, 256, 0, stream>>>(rel, anti);
    }

    dim3 gg(NROWS / 64, DD / 32);   // 64x32 tile per wave
    proj_gemm<<<gg, 32, 0, stream>>>(xpe, Wqb, bq, Qb, nullptr, 0);
    proj_gemm<<<gg, 32, 0, stream>>>(xpe, Wkb, bk, Kb, nullptr, 1);
    proj_gemm<<<gg, 32, 0, stream>>>(xpe, Wvb, bv, Vt, nullptr, 2);

    flash_attn<<<dim3(TT / 16, HH, BB), 32, 0, stream>>>(Qb, Kb, Vt, mask, anti, AO);

    proj_gemm<<<gg, 32, 0, stream>>>(AO, Wob, bo, nullptr, out, 3);
}
